// Attention_72791105732908
// MI455X (gfx1250) — compile-verified, matrix operand order checked by reading, not run on hardware
//
#include <hip/hip_runtime.h>
#include <hip/hip_bf16.h>
#include <math.h>
#include <stdint.h>

// ---------------------------------------------------------------------------
// L=2048, B=64, H=HC=1024 additive attention.
// Dominant cost: [131072 x 1024] @ [1024 x 1024] GEMM (275 GFLOP) in BF16 WMMA
// fused with +hb / tanh / W_v-dot epilogue.  B-matrix chunks are streamed into
// LDS by the Tensor Data Mover (double buffered, TENSORcnt-gated) so the WMMA
// pipeline never stalls on staging barriers.
// HBM traffic ~1 GB total => ~45us floor @ 23.3 TB/s; BF16 WMMA keeps the
// GEMM in the same ballpark instead of the >1ms an FP32 path would cost.
// ---------------------------------------------------------------------------

typedef __attribute__((ext_vector_type(16))) __bf16        bf16x16;
typedef __attribute__((ext_vector_type(8)))  float         v8f;
typedef __attribute__((ext_vector_type(4)))  unsigned int  u32x4;
typedef __attribute__((ext_vector_type(8)))  int           i32x8;
typedef __attribute__((ext_vector_type(4)))  int           i32x4;

#define L_DIM   2048
#define B_DIM   64
#define H_DIM   1024
#define HC_DIM  1024
#define NEG_BIG (-1e10f)

// LDS strides (elements). 32B-aligned, not a multiple of the 256B bank period.
#define A_STRIDE 1040   // 1024 + 16 pad  (2080B = 65*32)
#define B_STRIDE 272    // 256  + 16 pad  ( 544B = 17*32)

#if __has_builtin(__builtin_amdgcn_tensor_load_to_lds) && \
    __has_builtin(__builtin_amdgcn_s_wait_tensorcnt)
#define USE_TDM 1
#else
#define USE_TDM 0
#endif

__device__ __forceinline__ unsigned short f2bf(float f) {
    union { float f; unsigned u; } x; x.f = f;
    unsigned u = x.u;
    return (unsigned short)((u + 0x7FFFu + ((u >> 16) & 1u)) >> 16);  // RNE
}

// Hardware TANH transcendental on CDNA5 (V_TANH_F32); OCML fallback.
__device__ __forceinline__ float fast_tanh(float x) {
#if __has_builtin(__builtin_amdgcn_tanhf)
    return __builtin_amdgcn_tanhf(x);
#else
    return tanhf(x);
#endif
}

#if USE_TDM
// Issue one TDM load: 2-D tile of 128 rows x 256 bf16 out of the 1024x1024
// bf16 B matrix, row-padded by 16 bf16 (8 DWORDs) every 256 bf16 (128 DWORDs)
// so LDS rows land on the bank-conflict-free B_STRIDE layout.
__device__ __forceinline__ void tdm_load_B(const unsigned short* gptr,
                                           unsigned lds_off) {
    const unsigned long long ga = (unsigned long long)(uintptr_t)gptr;
    u32x4 g0;
    g0.x = 0x1u;                                   // count=1 (valid user D#)
    g0.y = lds_off;                                // lds_addr
    g0.z = (unsigned)(ga & 0xFFFFFFFFull);         // global_addr[31:0]
    g0.w = (unsigned)((ga >> 32) & 0x1FFFFFFull)   // global_addr[56:32]
         | (2u << 30);                             // type = 2 ("image")
    i32x8 g1;
    g1[0] = (int)((1u << 16)      // data_size = 2 bytes
                | (1u << 20)      // pad_enable
                | (6u << 22)      // pad_interval: 128 DWORDs (= one 512B row)
                | (7u << 25));    // pad_amount : 8 DWORDs  (= 32B row pad)
    g1[1] = (int)(((unsigned)H_DIM & 0xFFFFu) << 16);     // tensor_dim0 lo
    g1[2] = (int)(((unsigned)H_DIM >> 16)                 // tensor_dim0 hi
                | (((unsigned)HC_DIM & 0xFFFFu) << 16));  // tensor_dim1 lo
    g1[3] = (int)(((unsigned)HC_DIM >> 16)                // tensor_dim1 hi
                | (256u << 16));                          // tile_dim0 = 256
    g1[4] = (int)128u;                                    // tile_dim1 = 128
    g1[5] = (int)H_DIM;                                   // dim0_stride[31:0]
    g1[6] = 0;
    g1[7] = 0;
    const i32x4 z4 = {0, 0, 0, 0};
#if defined(__clang_major__) && __clang_major__ >= 23
    const i32x8 z8 = {0, 0, 0, 0, 0, 0, 0, 0};
    __builtin_amdgcn_tensor_load_to_lds(g0, g1, z4, z4, z8, 0);
#else
    __builtin_amdgcn_tensor_load_to_lds(g0, g1, z4, z4, 0);
#endif
}
#endif

// --------------------------------------------------------------------------
// K0: W_e[:, 1024:2048] fp32 -> bf16 [HC][H] (N-major: B frags contiguous).
// --------------------------------------------------------------------------
__global__ __launch_bounds__(256) void convert_we_kernel(
    const float* __restrict__ W_e, unsigned short* __restrict__ w2bf) {
    const int c = blockIdx.x;
    const int h = threadIdx.x * 4;
    float4 f = *(const float4*)(W_e + (size_t)c * (2 * H_DIM) + H_DIM + h);
    unsigned long long pk =
        (unsigned long long)f2bf(f.x)        |
        ((unsigned long long)f2bf(f.y) << 16) |
        ((unsigned long long)f2bf(f.z) << 32) |
        ((unsigned long long)f2bf(f.w) << 48);
    *(unsigned long long*)(w2bf + (size_t)c * H_DIM + h) = pk;
}

// --------------------------------------------------------------------------
// K1: hb[b,c] = dot(hidden[b,:], W_e[c,0:1024]) + b_e[c]   (tiny, L2-bound)
// --------------------------------------------------------------------------
__global__ __launch_bounds__(256) void hb_kernel(
    const float* __restrict__ hidden, const float* __restrict__ W_e,
    const float* __restrict__ b_e, float* __restrict__ hb) {
    const int lane = threadIdx.x & 31;
    const int w    = threadIdx.x >> 5;
    const int c    = blockIdx.x * 8 + w;
    const int b    = blockIdx.y;
    const float* we = W_e + (size_t)c * (2 * H_DIM);
    const float* hd = hidden + (size_t)b * H_DIM;
    float acc = 0.0f;
    for (int i = lane; i < H_DIM; i += 32) acc += hd[i] * we[i];
    acc += __shfl_xor(acc, 16);
    acc += __shfl_xor(acc, 8);
    acc += __shfl_xor(acc, 4);
    acc += __shfl_xor(acc, 2);
    acc += __shfl_xor(acc, 1);
    if (lane == 0) hb[b * HC_DIM + c] = acc + b_e[c];
}

// --------------------------------------------------------------------------
// K2: scores[b,l] = sum_c W_v[c]*tanh( outs[b,l,:] @ w2bf[c,:] + hb[b,c] )
// Block = 8 waves, 64 L-rows of one batch b.
//   A: 64x1024 bf16 staged once in LDS (fp32->bf16 on the fly).
//   B: 128-col x 256-K bf16 chunks, TDM double-buffered through LDS.
//   Wave w: 4 M-tiles x N-subtile [w*16, w*16+16), v_wmma_f32_16x16x32_bf16.
// --------------------------------------------------------------------------
__global__ __launch_bounds__(256) void fused_energy_score_kernel(
    const float* __restrict__ prev, const unsigned short* __restrict__ w2bf,
    const float* __restrict__ hb, const float* __restrict__ wv,
    float* __restrict__ scores) {
    __shared__ __align__(32) unsigned short ldsA[64 * A_STRIDE];        // 130 KB
    __shared__ __align__(32) unsigned short ldsB[2][128 * B_STRIDE];    // 136 KB
    __shared__ float lds_sc[64];

    const int tid  = threadIdx.x;
    const int lane = tid & 31;
    const int wid  = tid >> 5;     // 0..7
    const int hi   = lane >> 4;    // 0/1
    const int lo   = lane & 15;
    const int b      = blockIdx.y;
    const int l_base = blockIdx.x * 64;

    if (tid < 64) lds_sc[tid] = 0.0f;

#if USE_TDM
    // kick off chunk 0 before A staging so DMA overlaps the A conversion
    if (wid == 0) {
        tdm_load_B(w2bf, (unsigned)(uintptr_t)(void*)&ldsB[0][0]);
    }
#endif

    // ---- stage A: 64 rows x 1024 h, fp32 -> bf16 (coalesced 4KB per row) --
    for (int row = 0; row < 64; ++row) {
        const size_t goff = ((size_t)((l_base + row) * B_DIM + b)) * H_DIM + tid * 4;
        float4 f = *(const float4*)(prev + goff);
        unsigned long long pk =
            (unsigned long long)f2bf(f.x)        |
            ((unsigned long long)f2bf(f.y) << 16) |
            ((unsigned long long)f2bf(f.z) << 32) |
            ((unsigned long long)f2bf(f.w) << 48);
        *(unsigned long long*)(ldsA + row * A_STRIDE + tid * 4) = pk;
    }

    float part[32];
#pragma unroll
    for (int i = 0; i < 32; ++i) part[i] = 0.0f;

    v8f c0 = {}, c1 = {}, c2 = {}, c3 = {};

    // 32 chunks: q = n_outer*4 + kc ; chunk q covers cols [(q>>2)*128, +128)
    // and K range [(q&3)*256, +256).
    for (int q = 0; q < 32; ++q) {
        const int kc  = q & 3;
        const int buf = q & 1;
        if (kc == 0) {
            v8f zz = {};
            c0 = zz; c1 = zz; c2 = zz; c3 = zz;
        }

#if USE_TDM
        if (wid == 0) {
            if (q + 1 < 32) {
                const int qn = q + 1;
                tdm_load_B(w2bf + (size_t)(qn >> 2) * 128 * H_DIM + (qn & 3) * 256,
                           (unsigned)(uintptr_t)(void*)&ldsB[buf ^ 1][0]);
                __builtin_amdgcn_s_wait_tensorcnt(1);   // chunk q landed
            } else {
                __builtin_amdgcn_s_wait_tensorcnt(0);
            }
        }
        __syncthreads();                                // publish chunk q
#else
        __syncthreads();                                // chunk buf free
        {   // manual staging fallback: 16B per thread per pass
            const int cl = tid & 31;
            for (int rb = 0; rb < 128; rb += 8) {
                const int r = rb + wid;
                uint4 v = *(const uint4*)(w2bf + (size_t)((q >> 2) * 128 + r) * H_DIM
                                          + (q & 3) * 256 + cl * 8);
                *(uint4*)(&ldsB[buf][0] + r * B_STRIDE + cl * 8) = v;
            }
        }
        __syncthreads();
#endif

        const unsigned short* Bp = &ldsB[buf][0];
#pragma unroll
        for (int ki = 0; ki < 8; ++ki) {
            const int koff  = ki * 32 + hi * 16;
            const int akoff = kc * 256 + koff;
            bf16x16 bf = *(const bf16x16*)(Bp + (wid * 16 + lo) * B_STRIDE + koff);
            bf16x16 a0 = *(const bf16x16*)(ldsA + (0  + lo) * A_STRIDE + akoff);
            bf16x16 a1 = *(const bf16x16*)(ldsA + (16 + lo) * A_STRIDE + akoff);
            bf16x16 a2 = *(const bf16x16*)(ldsA + (32 + lo) * A_STRIDE + akoff);
            bf16x16 a3 = *(const bf16x16*)(ldsA + (48 + lo) * A_STRIDE + akoff);
            c0 = __builtin_amdgcn_wmma_f32_16x16x32_bf16(false, a0, false, bf,
                                                         (short)0, c0, false, false);
            c1 = __builtin_amdgcn_wmma_f32_16x16x32_bf16(false, a1, false, bf,
                                                         (short)0, c1, false, false);
            c2 = __builtin_amdgcn_wmma_f32_16x16x32_bf16(false, a2, false, bf,
                                                         (short)0, c2, false, false);
            c3 = __builtin_amdgcn_wmma_f32_16x16x32_bf16(false, a3, false, bf,
                                                         (short)0, c3, false, false);
        }
        __syncthreads();   // all waves done with buf before it is re-filled

        if (kc == 3) {
            // epilogue: +hb, tanh, *W_v (C layout: VGPR r -> M=r+8*hi, N=lo)
            const int   col = (q >> 2) * 128 + wid * 16 + lo;
            const float wvv = wv[col];
            const float hbv = hb[b * HC_DIM + col];
#pragma unroll
            for (int r = 0; r < 8; ++r) {
                part[0 * 8 + r] += wvv * fast_tanh(c0[r] + hbv);
                part[1 * 8 + r] += wvv * fast_tanh(c1[r] + hbv);
                part[2 * 8 + r] += wvv * fast_tanh(c2[r] + hbv);
                part[3 * 8 + r] += wvv * fast_tanh(c3[r] + hbv);
            }
        }
    }

    // reduce the 16 columns held by lanes sharing the same rows
#pragma unroll
    for (int m = 0; m < 32; ++m) {
        float v = part[m];
        v += __shfl_xor(v, 1);
        v += __shfl_xor(v, 2);
        v += __shfl_xor(v, 4);
        v += __shfl_xor(v, 8);
        if (lo == 0)
            atomicAdd(&lds_sc[(m >> 3) * 16 + hi * 8 + (m & 7)], v);  // ds_add_f32
    }
    __syncthreads();
    if (tid < 64) scores[b * L_DIM + l_base + tid] = lds_sc[tid];
}

// --------------------------------------------------------------------------
// K3: masked softmax over L per batch row.
// --------------------------------------------------------------------------
__global__ __launch_bounds__(256) void softmax_kernel(
    const float* __restrict__ scores, const int* __restrict__ mask,
    float* __restrict__ wout) {
    __shared__ float red[256];
    const int b = blockIdx.x, tid = threadIdx.x;
    float v[8];
#pragma unroll
    for (int i = 0; i < 8; ++i) {
        const int l = tid + i * 256;
        const float s = scores[b * L_DIM + l];
        v[i] = (mask[b * L_DIM + l] == 0) ? NEG_BIG : s;
    }
    float mx = v[0];
#pragma unroll
    for (int i = 1; i < 8; ++i) mx = fmaxf(mx, v[i]);
    red[tid] = mx; __syncthreads();
    for (int s = 128; s > 0; s >>= 1) {
        if (tid < s) red[tid] = fmaxf(red[tid], red[tid + s]);
        __syncthreads();
    }
    mx = red[0]; __syncthreads();
    float se = 0.0f;
#pragma unroll
    for (int i = 0; i < 8; ++i) se += __expf(v[i] - mx);
    red[tid] = se; __syncthreads();
    for (int s = 128; s > 0; s >>= 1) {
        if (tid < s) red[tid] += red[tid + s];
        __syncthreads();
    }
    const float inv = 1.0f / red[0];
#pragma unroll
    for (int i = 0; i < 8; ++i)
        wout[b * L_DIM + tid + i * 256] = __expf(v[i] - mx) * inv;
}

// --------------------------------------------------------------------------
// K4: out[b,h] = sum_l w[b,l] * prev[l,b,h]   (bandwidth bound, coalesced)
// --------------------------------------------------------------------------
__global__ __launch_bounds__(256) void weighted_sum_kernel(
    const float* __restrict__ prev, const float* __restrict__ wbuf,
    float* __restrict__ out) {
    __shared__ float wsh[L_DIM];
    const int tid = threadIdx.x;
    const int b   = blockIdx.y;
    const int h   = blockIdx.x * 256 + tid;
#pragma unroll
    for (int i = 0; i < 8; ++i)
        wsh[tid + i * 256] = wbuf[b * L_DIM + tid + i * 256];
    __syncthreads();
    float acc = 0.0f;
#pragma unroll 4
    for (int l = 0; l < L_DIM; ++l)
        acc += wsh[l] * prev[((size_t)l * B_DIM + b) * H_DIM + h];
    out[b * H_DIM + h] = acc;
}

// --------------------------------------------------------------------------
extern "C" void kernel_launch(void* const* d_in, const int* in_sizes, int n_in,
                              void* d_out, int out_size, void* d_ws, size_t ws_size,
                              hipStream_t stream) {
    (void)in_sizes; (void)n_in; (void)out_size; (void)ws_size;
    const float* prev   = (const float*)d_in[0];   // [L,B,H]
    const float* hidden = (const float*)d_in[1];   // [B,H]
    const int*   mask   = (const int*)  d_in[2];   // [B,L]
    const float* W_e    = (const float*)d_in[3];   // [HC,2H]
    const float* b_e    = (const float*)d_in[4];   // [HC]
    const float* W_v    = (const float*)d_in[5];   // [HC]
    float* out = (float*)d_out;                    // [1,B,H]

    char* ws = (char*)d_ws;
    unsigned short* w2bf = (unsigned short*)ws;                               // 2 MB
    float* hb     = (float*)(ws + (size_t)(2u << 20));                        // 256 KB
    float* scores = (float*)(ws + (size_t)(2u << 20) + (256u << 10));         // 512 KB
    float* wbuf   = (float*)(ws + (size_t)(2u << 20) + (768u << 10));         // 512 KB

    hipLaunchKernelGGL(convert_we_kernel, dim3(HC_DIM), dim3(256), 0, stream,
                       W_e, w2bf);
    hipLaunchKernelGGL(hb_kernel, dim3(HC_DIM / 8, B_DIM), dim3(256), 0, stream,
                       hidden, W_e, b_e, hb);
    hipLaunchKernelGGL(fused_energy_score_kernel, dim3(L_DIM / 64, B_DIM), dim3(256),
                       0, stream, prev, w2bf, hb, W_v, scores);
    hipLaunchKernelGGL(softmax_kernel, dim3(B_DIM), dim3(256), 0, stream,
                       scores, mask, wbuf);
    hipLaunchKernelGGL(weighted_sum_kernel, dim3(H_DIM / 256, B_DIM), dim3(256),
                       0, stream, prev, wbuf, out);
}
